// EPMixtralSparseMoeBlock_75058848464967
// MI455X (gfx1250) — matrix operand order repaired and checked by reading, not compiled, not run on hardware
//
#include <hip/hip_runtime.h>
#include <hip/hip_bf16.h>

// ---------------- problem constants ----------------
#define NE 8
#define TOPK 2
#define HID 1024
#define FFN 2048
#define T 2048          // B*S tokens

// LDS tile geometry for the expert kernel
#define XSTR 1032                        // bf16 elems per LDS row (1024 + 8 pad -> no bank aliasing)
#define XS_BYTES (32 * XSTR * 2)         // 66048 B per 32x1024 bf16 tile
#define SMEM_BYTES (2 * XS_BYTES + 256)  // xs + hmid + tok[32] + gate[32] = 132352 B (< 320KB WGP LDS)

typedef __attribute__((ext_vector_type(16))) __bf16 v16bf;
typedef __attribute__((ext_vector_type(8)))  float  v8f;

// ---------------- WMMA helpers ----------------
__device__ __forceinline__ v8f wmma_bf16(v16bf a, v16bf b, v8f c) {
  // D = A(16x32 bf16) x B(32x16 bf16) + C(16x16 f32)
  return __builtin_amdgcn_wmma_f32_16x16x32_bf16(false, a, false, b, (short)0, c,
                                                 false, false);
}

__device__ __forceinline__ v8f zero8() {
  v8f z;
#pragma unroll
  for (int i = 0; i < 8; ++i) z[i] = 0.0f;
  return z;
}

// fast sigmoid: single v_rcp_f32 instead of the IEEE divide expansion
__device__ __forceinline__ float fast_sigmoid(float g) {
  return __builtin_amdgcn_rcpf(1.0f + __expf(-g));
}

// A-fragment from an LDS row (16x32 A tile, this code's contiguous k order, same as B):
// lane L: m = L&15 (+16 for second M-tile handled by caller), h = L>>4
// elems 0..7  -> k = k0 + 16h + i      (one 16B chunk)
// elems 8..15 -> k = k0 + 16h + 8 + i  (one 16B chunk)
__device__ __forceinline__ v16bf lds_a_frag(const __bf16* row, int k0, int lh) {
  union { v16bf v; uint4 u[2]; } t;
  t.u[0] = *(const uint4*)(row + k0 + 16 * lh);
  t.u[1] = *(const uint4*)(row + k0 + 16 * lh + 8);
  return t.v;
}

// B-fragment: lane L holds column n = L&15; elems i -> k = k0 + 16*(L>>4) + i.
// Source is 16 consecutive f32 of a weight row, converted to bf16 in-register.
__device__ __forceinline__ v16bf glb_b_frag(const float* p) {
  const float4* q = (const float4*)p;
  float4 f0 = q[0], f1 = q[1], f2 = q[2], f3 = q[3];
  v16bf r;
  r[0]  = (__bf16)f0.x; r[1]  = (__bf16)f0.y; r[2]  = (__bf16)f0.z; r[3]  = (__bf16)f0.w;
  r[4]  = (__bf16)f1.x; r[5]  = (__bf16)f1.y; r[6]  = (__bf16)f1.z; r[7]  = (__bf16)f1.w;
  r[8]  = (__bf16)f2.x; r[9]  = (__bf16)f2.y; r[10] = (__bf16)f2.z; r[11] = (__bf16)f2.w;
  r[12] = (__bf16)f3.x; r[13] = (__bf16)f3.y; r[14] = (__bf16)f3.z; r[15] = (__bf16)f3.w;
  return r;
}

// ---------------- kernel 0: zero out + counters ----------------
__global__ void moe_init_kernel(float* __restrict__ out, int* __restrict__ counts) {
  size_t i = (size_t)blockIdx.x * blockDim.x + threadIdx.x;
  const size_t n4 = (size_t)T * HID / 4;
  if (i < n4) ((float4*)out)[i] = float4{0.f, 0.f, 0.f, 0.f};
  if (i < NE) counts[i] = 0;
}

// ---------------- kernel 1: router (one wave32 per token) ----------------
__global__ void moe_router_kernel(const float* __restrict__ x,
                                  const float* __restrict__ gw,
                                  float* __restrict__ logits,
                                  int* __restrict__ counts,
                                  int* __restrict__ tok_idx,
                                  float* __restrict__ tok_gate) {
  const int lane = threadIdx.x & 31;
  const int t = blockIdx.x * (blockDim.x >> 5) + (threadIdx.x >> 5);
  if (t >= T) return;

  float acc[NE];
#pragma unroll
  for (int e = 0; e < NE; ++e) acc[e] = 0.f;

  const float* xr = x + (size_t)t * HID;
  for (int j = lane; j < HID; j += 32) {
    float xv = xr[j];
#pragma unroll
    for (int e = 0; e < NE; ++e) acc[e] += xv * gw[e * HID + j];
  }
#pragma unroll
  for (int e = 0; e < NE; ++e) {
#pragma unroll
    for (int off = 16; off > 0; off >>= 1) acc[e] += __shfl_xor(acc[e], off, 32);
  }

  if (lane == 0) {
#pragma unroll
    for (int e = 0; e < NE; ++e) logits[(size_t)t * NE + e] = acc[e];
    // top-2 (stable ties like lax.top_k), renormalized softmax weights
    int i0 = 0;
#pragma unroll
    for (int e = 1; e < NE; ++e) if (acc[e] > acc[i0]) i0 = e;
    int i1 = (i0 == 0) ? 1 : 0;
#pragma unroll
    for (int e = 0; e < NE; ++e) if (e != i0 && acc[e] > acc[i1]) i1 = e;
    float p1 = __expf(acc[i1] - acc[i0]);                 // exp of (2nd - max)
    float w0 = __builtin_amdgcn_rcpf(1.f + p1);           // normalized top-2 weight
    int p = atomicAdd(&counts[i0], 1);
    tok_idx[i0 * T + p] = t;  tok_gate[i0 * T + p] = w0;
    p = atomicAdd(&counts[i1], 1);
    tok_idx[i1 * T + p] = t;  tok_gate[i1 * T + p] = 1.f - w0;
  }
}

// ---------------- kernel 2: fused SwiGLU expert FFN ----------------
// grid = (T/32 token tiles, NE experts), block = 256 (8 wave32)
// Stage A: hmid = silu(X w1^T) * (X w3^T)  -> bf16 LDS, half of FFN at a time
// Stage B: out_tile += hmid_half @ w2_half^T, accumulated in registers
__global__ __launch_bounds__(256, 1)
void moe_expert_kernel(const float* __restrict__ x,
                       const float* __restrict__ w1,
                       const float* __restrict__ w2,
                       const float* __restrict__ w3,
                       float* __restrict__ out,
                       const int* __restrict__ counts,
                       const int* __restrict__ tok_idx,
                       const float* __restrict__ tok_gate) {
  extern __shared__ char smem[];
  __bf16* xs   = (__bf16*)smem;                      // 32 x XSTR bf16 (X tile)
  __bf16* hm   = (__bf16*)(smem + XS_BYTES);         // 32 x XSTR bf16 (hmid half)
  int*    tokl = (int*)(smem + 2 * XS_BYTES);        // 32 token ids
  float*  gatl = (float*)(smem + 2 * XS_BYTES + 128);// 32 gate weights

  const int e    = blockIdx.y;
  const int cnt  = counts[e];
  const int row0 = blockIdx.x * 32;
  if (row0 >= cnt) return;
  const int rows = (cnt - row0 < 32) ? (cnt - row0) : 32;

  const int tid = threadIdx.x;
  if (tid < 32) {
    int m = row0 + tid;
    tokl[tid] = (m < cnt) ? tok_idx[e * T + m] : 0;
    gatl[tid] = (m < cnt) ? tok_gate[e * T + m] : 0.f;
  }
  __syncthreads();

  // gather X tile -> bf16 LDS (rows beyond count read token 0; gate==0 kills them)
  for (int c = tid; c < 32 * (HID / 4); c += 256) {
    int r  = c >> 8;              // 256 float4 chunks per row
    int c0 = (c & 255) * 4;
    float4 v = *(const float4*)(x + (size_t)tokl[r] * HID + c0);
    __bf16* d = xs + (size_t)r * XSTR + c0;
    d[0] = (__bf16)v.x; d[1] = (__bf16)v.y; d[2] = (__bf16)v.z; d[3] = (__bf16)v.w;
  }
  __syncthreads();

  const int wave = tid >> 5, lane = tid & 31;
  const int lm = lane & 15, lh = lane >> 4;

  v8f oacc[8][2];
#pragma unroll
  for (int i = 0; i < 8; ++i) { oacc[i][0] = zero8(); oacc[i][1] = zero8(); }

  const float* W1e = w1 + (size_t)e * FFN * HID;
  const float* W3e = w3 + (size_t)e * FFN * HID;
  const float* W2e = w2 + (size_t)e * HID * FFN;
  const __bf16* xr0 = xs + (size_t)lm * XSTR;
  const __bf16* xr1 = xs + (size_t)(lm + 16) * XSTR;
  const __bf16* hr0 = hm + (size_t)lm * XSTR;
  const __bf16* hr1 = hm + (size_t)(lm + 16) * XSTR;

  for (int fh = 0; fh < 2; ++fh) {
    // ---------- Stage A: 32 x 1024 hmid half ----------
    for (int ns = 0; ns < 8; ++ns) {
      const int nloc = wave * 128 + ns * 16 + lm;   // column within this half
      const int nglb = fh * 1024 + nloc;            // global FFN column
      const float* w1p = W1e + (size_t)nglb * HID + 16 * lh;
      const float* w3p = W3e + (size_t)nglb * HID + 16 * lh;
      v8f a1c0 = zero8(), a1c1 = zero8(), a3c0 = zero8(), a3c1 = zero8();
      for (int k0 = 0; k0 < HID; k0 += 32) {
        __builtin_prefetch(w1p + k0 + 256, 0, 0);   // global_prefetch on weight stream
        __builtin_prefetch(w3p + k0 + 256, 0, 0);
        v16bf a0 = lds_a_frag(xr0, k0, lh);
        v16bf a1 = lds_a_frag(xr1, k0, lh);
        v16bf b1 = glb_b_frag(w1p + k0);
        v16bf b3 = glb_b_frag(w3p + k0);
        a1c0 = wmma_bf16(a0, b1, a1c0);
        a1c1 = wmma_bf16(a1, b1, a1c1);
        a3c0 = wmma_bf16(a0, b3, a3c0);
        a3c1 = wmma_bf16(a1, b3, a3c1);
      }
      // SwiGLU fuse + store bf16 hmid (C/D layout: vgpr r -> m = 8h + r, n = lm)
#pragma unroll
      for (int r = 0; r < 8; ++r) {
        int m0 = 8 * lh + r;
        float g0 = a1c0[r];
        float v0 = g0 * fast_sigmoid(g0) * a3c0[r];
        hm[(size_t)m0 * XSTR + nloc] = (__bf16)v0;
        float g1 = a1c1[r];
        float v1 = g1 * fast_sigmoid(g1) * a3c1[r];
        hm[(size_t)(m0 + 16) * XSTR + nloc] = (__bf16)v1;
      }
    }
    __syncthreads();

    // ---------- Stage B: out_tile += hmid_half @ w2_half^T ----------
#pragma unroll
    for (int ns = 0; ns < 8; ++ns) {
      const int n = wave * 128 + ns * 16 + lm;      // hidden column
      const float* w2p = W2e + (size_t)n * FFN + fh * 1024 + 16 * lh;
      for (int k0 = 0; k0 < 1024; k0 += 32) {
        __builtin_prefetch(w2p + k0 + 256, 0, 0);
        v16bf a0 = lds_a_frag(hr0, k0, lh);
        v16bf a1 = lds_a_frag(hr1, k0, lh);
        v16bf b2 = glb_b_frag(w2p + k0);
        oacc[ns][0] = wmma_bf16(a0, b2, oacc[ns][0]);
        oacc[ns][1] = wmma_bf16(a1, b2, oacc[ns][1]);
      }
    }
    __syncthreads();   // hmid reused next half
  }

  // ---------- scatter: out[tok, n] += gate * acc ----------
#pragma unroll
  for (int ns = 0; ns < 8; ++ns) {
    const int n = wave * 128 + ns * 16 + lm;
#pragma unroll
    for (int mt = 0; mt < 2; ++mt) {
#pragma unroll
      for (int r = 0; r < 8; ++r) {
        int m = 16 * mt + 8 * lh + r;
        if (m < rows)
          atomicAdd(out + (size_t)tokl[m] * HID + n, gatl[m] * oacc[ns][mt][r]);
      }
    }
  }
}

// ---------------- host launcher ----------------
extern "C" void kernel_launch(void* const* d_in, const int* in_sizes, int n_in,
                              void* d_out, int out_size, void* d_ws, size_t ws_size,
                              hipStream_t stream) {
  const float* x  = (const float*)d_in[0];
  const float* gw = (const float*)d_in[1];
  const float* w1 = (const float*)d_in[2];
  const float* w2 = (const float*)d_in[3];
  const float* w3 = (const float*)d_in[4];

  float* out    = (float*)d_out;
  float* logits = out + (size_t)T * HID;   // second output, concatenated flat

  char*  ws       = (char*)d_ws;
  int*   counts   = (int*)ws;                                // 8 ints
  int*   tok_idx  = (int*)(ws + 256);                        // [NE][T]
  float* tok_gate = (float*)(ws + 256 + (size_t)NE * T * 4); // [NE][T]

  (void)hipFuncSetAttribute((const void*)moe_expert_kernel,
                            hipFuncAttributeMaxDynamicSharedMemorySize, SMEM_BYTES);

  // 1) zero output accumulator + expert counters (deterministic each call)
  moe_init_kernel<<<(T * HID / 4 + 255) / 256, 256, 0, stream>>>(out, counts);

  // 2) router: logits, top-2 gates, per-expert token lists
  moe_router_kernel<<<T / 8, 256, 0, stream>>>(x, gw, logits, counts, tok_idx, tok_gate);

  // 3) fused SwiGLU expert FFN with WMMA + scatter-add
  moe_expert_kernel<<<dim3(T / 32, NE), 256, SMEM_BYTES, stream>>>(
      x, w1, w2, w3, out, counts, tok_idx, tok_gate);
}
